// Dynamics_40510131536026
// MI455X (gfx1250) — matrix operand order repaired and checked by reading, not compiled, not run on hardware
//
#include <hip/hip_runtime.h>
#include <hip/hip_bf16.h>
#include <math.h>

// ---- CDNA5 WMMA vector types -------------------------------------------------
typedef _Float16 half16 __attribute__((ext_vector_type(16)));
typedef float    float8 __attribute__((ext_vector_type(8)));
typedef float    float2v __attribute__((ext_vector_type(2)));

#define T_STEPS 64   // encoder length (reference: T = 64)
#define ROWS_PER_WAVE 16
#define WAVES_PER_BLOCK 8
#define ROWS_PER_BLOCK (ROWS_PER_WAVE * WAVES_PER_BLOCK) // 128

// ---- Fast activations --------------------------------------------------------
// gfx1250 has a hardware v_tanh_f32 (tanh-insts). Use it if clang declares the
// builtin; otherwise fall back to the exp2/rcp formulation.
#if __has_builtin(__builtin_amdgcn_tanhf)
__device__ __forceinline__ float tanh_fast(float x) {
    return __builtin_amdgcn_tanhf(x);              // 1x v_tanh_f32
}
__device__ __forceinline__ float sigm(float x) {
    // sigmoid(x) = 0.5*tanh(x/2) + 0.5   (mul + tanh + fma)
    return __builtin_fmaf(0.5f, __builtin_amdgcn_tanhf(0.5f * x), 0.5f);
}
#else
__device__ __forceinline__ float tanh_fast(float x) {
    return 1.0f - 2.0f * __builtin_amdgcn_rcpf(1.0f + __builtin_amdgcn_exp2f(2.88539008f * x));
}
__device__ __forceinline__ float sigm(float x) {
    return __builtin_amdgcn_rcpf(1.0f + __builtin_amdgcn_exp2f(-1.44269504f * x));
}
#endif

// Wave-local LDS fence: same-wave DS ops are in-order; this orders them vs the
// compiler and drains DScnt before the dependent ds_load.
__device__ __forceinline__ void lds_fence() {
    asm volatile("s_wait_dscnt 0" ::: "memory");
}

// K-permutation pi: hardware K slot -> logical hidden column j.
// Derived so that the C-layout -> A-layout hand-off through LDS is a single
// packed b32 store per (lane, r). Applied identically to the B (weight)
// fragments, which makes the permutation semantically invisible.
__device__ __forceinline__ int kperm(int k) {
    return 8 * (k >> 4) + ((k >> 1) & 7) + 16 * (k & 1);
}

// LDS slot (in dwords) for the packed pair (h[m][q], h[m][q+16]) produced by
// C-layout lane `lane`, register r.  lane' = r + 8*hl + 16*(q>=8), v = q&7.
__device__ __forceinline__ int a_store_idx(int lane, int r) {
    int lp = r + ((lane >> 4) << 3) + ((((lane & 15) >> 3) & 1) << 4);
    return (lp << 3) | (lane & 7);
}

__device__ __forceinline__ unsigned pack_h2(float a, float b) {
    unsigned short ua = __builtin_bit_cast(unsigned short, (_Float16)a);
    unsigned short ub = __builtin_bit_cast(unsigned short, (_Float16)b);
    return (unsigned)ua | ((unsigned)ub << 16);
}

// One LSTM step for a 16-row batch tile owned by this wave.
// gates(16x128) = x(16x4)@Wih.T  (v_wmma_f32_16x16x4_f32, exact f32)
//              + h(16x32)@Whh.T  (v_wmma_f32_16x16x32_f16, f16 in / f32 acc)
// Returns the new h as an A-fragment (read back through wave-private LDS).
__device__ __forceinline__ half16 lstm_step(
    float2v ax, half16 ah, const float2v* bih, const half16* bhh,
    float8* creg, unsigned* hrow, int lane)
{
    float8 z = {};
    float8 acc[8];
#pragma unroll
    for (int n = 0; n < 8; ++n) {
        float8 xi = __builtin_amdgcn_wmma_f32_16x16x4_f32(
            false, ax, false, bih[n], (short)0, z, false, false);
        acc[n] = __builtin_amdgcn_wmma_f32_16x16x32_f16(
            false, ah, false, bhh[n], (short)0, xi, false, false);
    }
    // Elementwise gate math in C-layout registers; pack new h into LDS in the
    // permuted A layout (one b32 store per r).
#pragma unroll
    for (int r = 0; r < 8; ++r) {
        float h0 = 0.0f, h1 = 0.0f;
#pragma unroll
        for (int t = 0; t < 2; ++t) {
            float iv = sigm(acc[t][r]);
            float fv = sigm(acc[2 + t][r]);
            float gv = tanh_fast(acc[4 + t][r]);
            float ov = sigm(acc[6 + t][r]);
            float cn = fv * creg[t][r] + iv * gv;
            creg[t][r] = cn;
            float hv = ov * tanh_fast(cn);
            if (t == 0) h0 = hv; else h1 = hv;
        }
        hrow[a_store_idx(lane, r)] = pack_h2(h0, h1);
    }
    lds_fence();
    return *reinterpret_cast<const half16*>(&hrow[lane << 3]); // 32B/lane
}

__global__ __launch_bounds__(256) void dynamics_lstm_kernel(
    const float* __restrict__ poses, const float* __restrict__ Wih,
    const float* __restrict__ Whh, const float* __restrict__ Wd,
    const int* __restrict__ n_new_ptr, float* __restrict__ out, int B)
{
    __shared__ __align__(32) unsigned hstage[WAVES_PER_BLOCK][256]; // 16x32 f16 tile per wave
    __shared__ float velstage[WAVES_PER_BLOCK][32];                 // cs_raw[16] | sn_raw[16]

    const int tid  = threadIdx.x;
    const int w    = tid >> 5;
    const int lane = tid & 31;
    const int col  = lane & 15;     // N position / batch row within tile
    const int hl   = lane >> 4;     // lane half
    const int row  = blockIdx.x * ROWS_PER_BLOCK + w * ROWS_PER_WAVE + col;

    unsigned* hrow   = &hstage[w][0];
    float*    velrow = &velstage[w][0];

    // ---- one-time weight fragment construction (pi-permuted K) -------------
    half16 bhh[8];
#pragma unroll
    for (int n = 0; n < 8; ++n) {
#pragma unroll
        for (int v = 0; v < 8; ++v) {
#pragma unroll
            for (int p = 0; p < 2; ++p) {
                int k = 16 * hl + 2 * v + p;
                bhh[n][2 * v + p] = (_Float16)Whh[(16 * n + col) * 32 + kperm(k)];
            }
        }
    }
    half16 bden;
#pragma unroll
    for (int v = 0; v < 8; ++v) {
#pragma unroll
        for (int p = 0; p < 2; ++p) {
            int k = 16 * hl + 2 * v + p;
            bden[2 * v + p] = (col < 3) ? (_Float16)Wd[col * 32 + kperm(k)]
                                        : (_Float16)0.0f;
        }
    }
    float2v bih[8];
#pragma unroll
    for (int n = 0; n < 8; ++n) {
        int k0 = 2 * hl; // lanes 0-15: K=0,1 ; lanes 16-31: K=2,3
        bih[n][0] = Wih[(16 * n + col) * 4 + k0];
        bih[n][1] = Wih[(16 * n + col) * 4 + k0 + 1];
    }

    // last encoder pose (x,y persist through the whole decoder)
    const float4 lastp =
        *reinterpret_cast<const float4*>(poses + ((size_t)(T_STEPS - 1) * B + row) * 4);

    // ---- encoder -----------------------------------------------------------
    half16 ah = {};            // h0 = 0 (A fragment)
    float8 creg[2];            // c state in C layout
    creg[0] = (float8){};
    creg[1] = (float8){};

    for (int t = 0; t < T_STEPS; ++t) {
        const float4 p =
            *reinterpret_cast<const float4*>(poses + ((size_t)t * B + row) * 4);
        if (t + 1 < T_STEPS)
            __builtin_prefetch(poses + ((size_t)(t + 1) * B + row) * 4, 0, 1);
        float2v ax;
        ax[0] = hl ? p.z : p.x;  // A f32 16x4: lanes0-15 K=0,1 ; lanes16-31 K=2,3
        ax[1] = hl ? p.w : p.y;
        ah = lstm_step(ax, ah, bih, bhh, creg, hrow, lane);
    }

    // ---- last_velocity = h @ Wd.T  (cols 0..2 valid) -----------------------
    const int n_dec = *n_new_ptr;
    float8 z = {};
    float8 vel = __builtin_amdgcn_wmma_f32_16x16x32_f16(
        false, ah, false, bden, (short)0, z, false, false);
    {
        int mb = hl << 3;
        if (col == 1) {
#pragma unroll
            for (int r = 0; r < 8; ++r) velrow[mb + r] = vel[r];      // cs_raw
        } else if (col == 2) {
#pragma unroll
            for (int r = 0; r < 8; ++r) velrow[16 + mb + r] = vel[r]; // sn_raw
        }
        lds_fence();
    }

    // ---- decoder -----------------------------------------------------------
    const float px = lastp.x, py = lastp.y;
    for (int t = 0; t < n_dec; ++t) {
        float c1 = velrow[col];
        float c2 = velrow[16 + col];
        float rn = __builtin_amdgcn_rsqf(c1 * c1 + c2 * c2); // v_rsq_f32
        float cs = c1 * rn, sn = c2 * rn;

        if (lane < 16) { // one float4 store per batch row
            float4 op;
            op.x = px; op.y = py; op.z = cs; op.w = sn;
            *reinterpret_cast<float4*>(out + ((size_t)t * B + row) * 4) = op;
        }

        float2v ax;
        ax[0] = hl ? cs : px;
        ax[1] = hl ? sn : py;
        ah = lstm_step(ax, ah, bih, bhh, creg, hrow, lane);

        vel = __builtin_amdgcn_wmma_f32_16x16x32_f16(
            false, ah, false, bden, (short)0, z, false, false);
        int mb = hl << 3;
        if (col == 1) {
#pragma unroll
            for (int r = 0; r < 8; ++r) velrow[mb + r] = vel[r];
        } else if (col == 2) {
#pragma unroll
            for (int r = 0; r < 8; ++r) velrow[16 + mb + r] = vel[r];
        }
        lds_fence();
    }
}

extern "C" void kernel_launch(void* const* d_in, const int* in_sizes, int n_in,
                              void* d_out, int out_size, void* d_ws, size_t ws_size,
                              hipStream_t stream) {
    (void)n_in; (void)d_ws; (void)ws_size; (void)out_size;
    const float* poses = (const float*)d_in[0];
    // d_in[1] = deltas: values are never used by the reference (only its length,
    // which equals n_new_poses = d_in[5]).
    const float* Wih = (const float*)d_in[2];
    const float* Whh = (const float*)d_in[3];
    const float* Wd  = (const float*)d_in[4];
    const int* nnp   = (const int*)d_in[5];
    float* out       = (float*)d_out;

    const int B = in_sizes[1] / T_STEPS;     // deltas has T*B elements
    dim3 grid((B + ROWS_PER_BLOCK - 1) / ROWS_PER_BLOCK);
    dim3 block(256);
    dynamics_lstm_kernel<<<grid, block, 0, stream>>>(poses, Wih, Whh, Wd, nnp, out, B);
}
